// PytorchMambaBlock_17257178596059
// MI455X (gfx1250) — compile-verified, matrix operand order checked by reading, not run on hardware
//
#include <hip/hip_runtime.h>
#include <math.h>

// ---------------------------------------------------------------------------
// Mamba block for MI455X (gfx1250): bf16 WMMA GEMMs + fp32 fused scan.
// D_MODEL=2048, D_INNER=4096, D_STATE=16, D_CONV=4, DT_RANK=128, B=2, L=1024
// ---------------------------------------------------------------------------

typedef __bf16 bf16_t;
typedef __attribute__((ext_vector_type(16))) __bf16 v16bf;
typedef __attribute__((ext_vector_type(8)))  float  v8f;

#define ROWS   2048      // B*L
#define DMODEL 2048
#define DINNER 4096
#define DSTATE 16
#define DTRANK 128
#define SEQL   1024
#define NBATCH 2

#define TM 128   // block tile M
#define TN 32    // block tile N (2 x 16 per wave)
#define TK 64    // K per LDS stage (2 x 32 WMMA k-steps)
#define LDP 68   // LDS row stride in bf16 (multiple of 4 -> 8B-aligned stores)

// K index held by (lane, element e) for the 16-bit A/B fragment layout
// (ISA 7.12.2: lanes 0-15 hold K {0..7,16..23}, lanes 16-31 hold K {8..15,24..31})
__device__ __forceinline__ int frag_k(int lane, int e) {
    int grp = (lane >> 4) & 1;
    return e + grp * 8 + ((e >= 8) ? 8 : 0);
}

__device__ __forceinline__ float silu_f(float x) {
    return x / (1.0f + __expf(-x));
}

// pack two f32 -> packed bf16x2 (v_cvt_pk_bf16_f32)
__device__ __forceinline__ unsigned pk_bf16(float lo, float hi) {
    __bf16 a = (__bf16)lo, b = (__bf16)hi;
    unsigned short ua = __builtin_bit_cast(unsigned short, a);
    unsigned short ub = __builtin_bit_cast(unsigned short, b);
    return (unsigned)ua | ((unsigned)ub << 16);
}

// C[M x N] = act( A[M x K](lda) * W[N x K]^T + bias )
// A, W fp32 in memory; converted to bf16 while staging into LDS (b128 loads,
// packed b64 LDS stores). act: 0 = none, 1 = softplus
__global__ __launch_bounds__(256)
void gemm_bf16_wmma(const float* __restrict__ A,
                    const float* __restrict__ W,
                    const float* __restrict__ bias,
                    float* __restrict__ C,
                    int M, int N, int K, int lda, int act)
{
    __shared__ __bf16 As[TM][LDP];
    __shared__ __bf16 Bs[TN][LDP];

    const int m0   = blockIdx.x * TM;
    const int n0   = blockIdx.y * TN;
    const int tid  = threadIdx.x;
    const int lane = tid & 31;
    const int wave = tid >> 5;            // 8 waves -> 8 M-subtiles of 16

    // ---- per-thread staging coordinates (hoisted out of the K loop) ----
    // A tile: 128x64 f32 = 2048 float4, 8 per thread (t adds 16 rows)
    const int ar0 = tid >> 4;             // 0..15
    const int ac4 = (tid & 15) << 2;      // float4 column * 4
    const float* aRow[8];
#pragma unroll
    for (int t = 0; t < 8; ++t)
        aRow[t] = A + (long)(m0 + ar0 + t * 16) * lda + ac4;
    // B tile: 32x64 f32 = 512 float4, 2 per thread
    const int br0 = tid >> 4;             // 0..15 (+16 for t=1)
    const float* bRow[2];
#pragma unroll
    for (int t = 0; t < 2; ++t)
        bRow[t] = W + (long)(n0 + br0 + t * 16) * K + ac4;

    v8f acc0 = {}, acc1 = {};

    for (int k0 = 0; k0 < K; k0 += TK) {
        // ---- issue all global loads for this stage (pipelined b128) ----
        float4 av[8], bv[2];
#pragma unroll
        for (int t = 0; t < 8; ++t)
            av[t] = *(const float4*)(aRow[t] + k0);
#pragma unroll
        for (int t = 0; t < 2; ++t)
            bv[t] = *(const float4*)(bRow[t] + k0);

        // prefetch next stage into L2 (global_prefetch_b8)
        if (k0 + TK < K) {
            __builtin_prefetch(aRow[tid & 7] + k0 + TK, 0, 0);
            __builtin_prefetch(bRow[tid & 1] + k0 + TK, 0, 0);
        }
        __syncthreads();   // protect LDS from previous stage's readers

        // ---- convert + packed 8B stores into LDS ----
#pragma unroll
        for (int t = 0; t < 8; ++t) {
            uint2 p = make_uint2(pk_bf16(av[t].x, av[t].y), pk_bf16(av[t].z, av[t].w));
            *(uint2*)&As[ar0 + t * 16][ac4] = p;
        }
#pragma unroll
        for (int t = 0; t < 2; ++t) {
            uint2 p = make_uint2(pk_bf16(bv[t].x, bv[t].y), pk_bf16(bv[t].z, bv[t].w));
            *(uint2*)&Bs[br0 + t * 16][ac4] = p;
        }
        __syncthreads();

        // ---- 2 k-halves x (1 A-frag x 2 B-frags) = 4 WMMAs per stage ----
        const int am = lane & 15;
#pragma unroll
        for (int h = 0; h < 2; ++h) {
            v16bf af, bf0, bf1;
#pragma unroll
            for (int e = 0; e < 16; ++e) {
                int k = h * 32 + frag_k(lane, e);
                af[e]  = As[wave * 16 + am][k];
                bf0[e] = Bs[am][k];
                bf1[e] = Bs[16 + am][k];
            }
            acc0 = __builtin_amdgcn_wmma_f32_16x16x32_bf16(
                false, af, false, bf0, (short)0, acc0, false, false);
            acc1 = __builtin_amdgcn_wmma_f32_16x16x32_bf16(
                false, af, false, bf1, (short)0, acc1, false, false);
        }
    }

    // ---- epilogue: VGPR r holds row r (lanes 0-15) / r+8 (lanes 16-31) ----
    const int hi = lane >> 4;
    const int nl = lane & 15;
#pragma unroll
    for (int nsub = 0; nsub < 2; ++nsub) {
        const int col = n0 + nsub * 16 + nl;
        const float bvs = bias ? bias[col] : 0.0f;
        const v8f& acc = nsub ? acc1 : acc0;
#pragma unroll
        for (int r = 0; r < 8; ++r) {
            int row = m0 + wave * 16 + r + hi * 8;
            float v = acc[r] + bvs;
            if (act == 1) v = (v > 20.0f) ? v : log1pf(__expf(v));
            C[(long)row * N + col] = v;
        }
    }
}

// xc = silu(depthwise_conv4(xp) + conv_b); xp = xz[..., :DINNER]
__global__ __launch_bounds__(256)
void conv_silu_kernel(const float* __restrict__ xz,
                      const float* __restrict__ conv_w,
                      const float* __restrict__ conv_b,
                      float* __restrict__ xc)
{
    int idx = blockIdx.x * 256 + threadIdx.x;     // over ROWS*DINNER
    if (idx >= ROWS * DINNER) return;
    int d  = idx & (DINNER - 1);
    int bl = idx >> 12;                            // /4096
    int l  = bl & (SEQL - 1);
    int b  = bl >> 10;

    float acc = conv_b[d];
#pragma unroll
    for (int k = 0; k < 4; ++k) {
        int ll = l + k - 3;
        if (ll >= 0)
            acc += xz[((long)(b * SEQL + ll) * (2 * DINNER)) + d] * conv_w[d * 4 + k];
    }
    xc[idx] = silu_f(acc);
}

// Selective scan, fused with y = (sum_s h*C + xc*D) * silu(z).
// One channel per thread; 16 states live in registers; serial over L.
__global__ __launch_bounds__(256)
void scan_kernel(const float* __restrict__ xz,     // for z = xz[..., DINNER:]
                 const float* __restrict__ xc,
                 const float* __restrict__ dbc,    // [ROWS][160]: dt|B|C
                 const float* __restrict__ delta,  // [ROWS][DINNER]
                 const float* __restrict__ A_log,
                 const float* __restrict__ Dp,
                 float* __restrict__ y)
{
    const int d = blockIdx.x * 256 + threadIdx.x;  // 0..4095
    const int b = blockIdx.y;

    float Acoef[DSTATE], h[DSTATE];
#pragma unroll
    for (int s = 0; s < DSTATE; ++s) {
        Acoef[s] = -__expf(A_log[d * DSTATE + s]);
        h[s] = 0.0f;
    }
    const float Dpd = Dp[d];

    for (int l = 0; l < SEQL; ++l) {
        const long bl  = (long)b * SEQL + l;
        const float dlt = delta[bl * DINNER + d];
        const float xcv = xc[bl * DINNER + d];
        const float zv  = xz[bl * (2 * DINNER) + DINNER + d];
        const float* bc = dbc + bl * (DTRANK + 2 * DSTATE) + DTRANK;  // B then C

        float accy = 0.0f;
#pragma unroll
        for (int s = 0; s < DSTATE; ++s) {
            float Bm = bc[s];
            float Cm = bc[DSTATE + s];
            float dA = __expf(dlt * Acoef[s]);
            h[s] = dA * h[s] + dlt * Bm * xcv;
            accy += h[s] * Cm;
        }
        y[bl * DINNER + d] = (accy + xcv * Dpd) * silu_f(zv);
    }
}

extern "C" void kernel_launch(void* const* d_in, const int* in_sizes, int n_in,
                              void* d_out, int out_size, void* d_ws, size_t ws_size,
                              hipStream_t stream)
{
    const float* x      = (const float*)d_in[0];
    const float* W_in   = (const float*)d_in[1];
    const float* conv_w = (const float*)d_in[2];
    const float* conv_b = (const float*)d_in[3];
    const float* W_x    = (const float*)d_in[4];
    const float* W_dt   = (const float*)d_in[5];
    const float* b_dt   = (const float*)d_in[6];
    const float* A_log  = (const float*)d_in[7];
    const float* Dp     = (const float*)d_in[8];
    const float* W_out  = (const float*)d_in[9];
    float* out = (float*)d_out;

    char* ws = (char*)d_ws;
    float* xz    = (float*)(ws);                                  // 2048*8192 f32
    float* xc    = (float*)(ws + (long)ROWS * 2 * DINNER * 4);    // 2048*4096
    float* dbc   = (float*)((char*)xc + (long)ROWS * DINNER * 4); // 2048*160
    float* delta = (float*)((char*)dbc + (long)ROWS * 160 * 4);   // 2048*4096
    float* yb    = (float*)((char*)delta + (long)ROWS * DINNER * 4);

    dim3 blk(256);

    // 1) xz = x @ W_in^T   (M=2048, N=8192, K=2048)
    gemm_bf16_wmma<<<dim3(ROWS / TM, (2 * DINNER) / TN), blk, 0, stream>>>(
        x, W_in, nullptr, xz, ROWS, 2 * DINNER, DMODEL, DMODEL, 0);

    // 2) xc = silu(conv(xp) + b)
    conv_silu_kernel<<<dim3((ROWS * DINNER) / 256), blk, 0, stream>>>(
        xz, conv_w, conv_b, xc);

    // 3) dbc = xc @ W_x^T  (M=2048, N=160, K=4096)
    gemm_bf16_wmma<<<dim3(ROWS / TM, 160 / TN), blk, 0, stream>>>(
        xc, W_x, nullptr, dbc, ROWS, 160, DINNER, DINNER, 0);

    // 4) delta = softplus(dt @ W_dt^T + b_dt)  (M=2048, N=4096, K=128, lda=160)
    gemm_bf16_wmma<<<dim3(ROWS / TM, DINNER / TN), blk, 0, stream>>>(
        dbc, W_dt, b_dt, delta, ROWS, DINNER, DTRANK, 160, 1);

    // 5) selective scan + output gating
    scan_kernel<<<dim3(DINNER / 256, NBATCH), blk, 0, stream>>>(
        xz, xc, dbc, delta, A_log, Dp, yb);

    // 6) out = y @ W_out^T (M=2048, N=2048, K=4096)
    gemm_bf16_wmma<<<dim3(ROWS / TM, DMODEL / TN), blk, 0, stream>>>(
        yb, W_out, nullptr, out, ROWS, DMODEL, DINNER, DINNER, 0);
}
